// NeighborhoodAttn_27882927686305
// MI455X (gfx1250) — matrix operand order repaired and checked by reading, not compiled, not run on hardware
//
#include <hip/hip_runtime.h>

typedef __attribute__((ext_vector_type(16))) __bf16 v16bf;
typedef __attribute__((ext_vector_type(8)))  float  v8f;
typedef __attribute__((ext_vector_type(4)))  unsigned int u32x4;
typedef __attribute__((ext_vector_type(4)))  int i32x4;
typedef __attribute__((ext_vector_type(8)))  int i32x8;

#define NB    2
#define DIMC  512
#define HH    48
#define WW    48
#define TT    (HH*WW)     /* 2304 tokens per image */
#define HEADS 8
#define HD    64
#define QK_SCALE 0.125f   /* 64^-0.5 */

#if __has_builtin(__builtin_amdgcn_tensor_load_to_lds) && \
    __has_builtin(__builtin_amdgcn_s_wait_tensorcnt)
#define USE_TDM 1
#else
#define USE_TDM 0
#endif

// ---------------------------------------------------------------------------
// helpers
// ---------------------------------------------------------------------------
__device__ inline v8f wmma_bf16(v16bf a, v16bf b, v8f c) {
  // D = A(16x32 bf16) * B(32x16 bf16) + C(16x16 f32)
  return __builtin_amdgcn_wmma_f32_16x16x32_bf16(false, a, false, b, (short)0, c,
                                                 false, false);
}

// build a 16-element bf16 fragment from two 16-byte chunks
__device__ inline v16bf frag16(const __bf16* p0, const __bf16* p1) {
  union { u32x4 u[2]; v16bf v; } r;
  r.u[0] = *(const u32x4*)p0;
  r.u[1] = *(const u32x4*)p1;
  return r.v;
}

__device__ inline int clampi(int v, int lo, int hi) {
  return v < lo ? lo : (v > hi ? hi : v);
}

#if USE_TDM
// byte offset of a __shared__ object within LDS
__device__ inline unsigned lds_addr_of(const void* p) {
  return (unsigned)(unsigned long long)(__attribute__((address_space(3))) const void*)p;
}

// Issue a TDM 2D tile load: tile_x (contiguous) x tile_y rows of bf16,
// global row stride = row_stride elements, into LDS at lds_byte_off
// (packed row-major, 2*tile_x bytes per row).
__device__ inline void tdm_load_tile_bf16(const __bf16* gptr, unsigned lds_byte_off,
                                          unsigned tile_x, unsigned tile_y,
                                          unsigned row_stride,
                                          unsigned tdim0, unsigned tdim1) {
  unsigned long long ga = (unsigned long long)(const void*)gptr;
  // D# group 0: count=1 (valid user descriptor), lds_addr, global_addr, type=2
  u32x4 g0 = { 1u,
               lds_byte_off,
               (unsigned)(ga & 0xFFFFFFFFu),
               (unsigned)((ga >> 32) & 0x1FFFFFFu) | (2u << 30) };
  // D# group 1: data_size=1 (2 bytes), tensor dims, tile dims, dim0 stride
  i32x8 g1;
  g1[0] = (int)(1u << 16);                                   // data_size = 2B
  g1[1] = (int)((tdim0 & 0xFFFFu) << 16);                    // tensor_dim0 lo16
  g1[2] = (int)(((tdim0 >> 16) & 0xFFFFu) |
                ((tdim1 & 0xFFFFu) << 16));                  // dim0 hi | dim1 lo
  g1[3] = (int)(((tdim1 >> 16) & 0xFFFFu) |
                ((tile_x & 0xFFFFu) << 16));                 // dim1 hi | tile_dim0
  g1[4] = (int)(tile_y & 0xFFFFu);                           // tile_dim1 (dim2=0)
  g1[5] = (int)row_stride;                                   // tensor_dim0_stride
  g1[6] = 0;
  g1[7] = 0;
  i32x4 gz  = {0, 0, 0, 0};
  i32x8 gz8 = {0, 0, 0, 0, 0, 0, 0, 0};
#if defined(__clang_major__) && __clang_major__ >= 23
  __builtin_amdgcn_tensor_load_to_lds(g0, g1, gz, gz, gz8, 0);
#else
  __builtin_amdgcn_tensor_load_to_lds(g0, g1, gz, gz, 0);
#endif
}
#endif  // USE_TDM

// ---------------------------------------------------------------------------
// fp32 -> bf16 conversion
// ---------------------------------------------------------------------------
__global__ __launch_bounds__(256)
void cvt_bf16_kernel(const float* __restrict__ in, __bf16* __restrict__ out, int n) {
  int i = (blockIdx.x * 256 + threadIdx.x) * 4;
  if (i + 3 < n) {
    float4 f = *(const float4*)(in + i);
    out[i + 0] = (__bf16)f.x;
    out[i + 1] = (__bf16)f.y;
    out[i + 2] = (__bf16)f.z;
    out[i + 3] = (__bf16)f.w;
  } else {
    for (; i < n; ++i) out[i] = (__bf16)in[i];
  }
}

// ---------------------------------------------------------------------------
// 1x1-conv projection as WMMA GEMM:  out[o,p] = sum_c W[o,c] * X[c,p] + bias[o]
// X layout [b][c][t] (t contiguous).  Block tile 128(M) x 64(N), K-step 32.
// W tiles staged with the Tensor Data Mover (double buffered, TENSORcnt);
// X tiles staged manually (transpose required).
// mode 0: write bf16 in [b][head][t][hd] layout, scaled (for q/k/v)
// mode 1: write fp32 in [b][c][h][w] layout (final output)
// ---------------------------------------------------------------------------
__global__ __launch_bounds__(256)
void proj_gemm_kernel(const __bf16* __restrict__ Wb,     // [512,512] row-major
                      const float*  __restrict__ bias,   // [512]
                      const __bf16* __restrict__ Xb,     // [B*512, 2304]
                      __bf16* __restrict__ out_qkv,
                      float*  __restrict__ out_f32,
                      float scale, int mode)
{
  __shared__ __align__(16) __bf16 Wl[2][128][32];  // A tiles, [m][k], dbl-buffer
  __shared__ __align__(16) __bf16 Xl[64][32];      // B tile, stored [n][k]

  const int tid   = threadIdx.x;
  const int lane  = tid & 31;
  const int wave  = tid >> 5;
  const int mbase = blockIdx.y * 128;
  const int ntile = blockIdx.x;          // 0..71 (36 tiles of 64 cols per batch)
  const int bi    = ntile / 36;
  const int tb    = (ntile % 36) * 64;

  const int wm   = (wave >> 1) * 32;     // wave row offset in block tile
  const int wn   = (wave & 1) * 32;      // wave col offset
  const int nl   = lane & 15;
  const int half = lane >> 4;

  const v8f vzero = {0.f,0.f,0.f,0.f,0.f,0.f,0.f,0.f};
  v8f acc[2][2];
  acc[0][0] = vzero; acc[0][1] = vzero; acc[1][0] = vzero; acc[1][1] = vzero;

#if USE_TDM
  unsigned woff[2];
  woff[0] = lds_addr_of(&Wl[0][0][0]);
  woff[1] = lds_addr_of(&Wl[1][0][0]);
  if (wave == 0) {  // prologue: DMA first W tile into buffer 0
    tdm_load_tile_bf16(Wb + (size_t)mbase * 512, woff[0], 32, 128, 512, 512, 512);
  }
#endif

  int cur = 0;
  for (int kt = 0; kt < 512; kt += 32, cur ^= 1) {
#if USE_TDM
    if (wave == 0 && kt + 32 < 512) {  // DMA next W tile into alternate buffer
      tdm_load_tile_bf16(Wb + (size_t)mbase * 512 + (kt + 32), woff[cur ^ 1],
                         32, 128, 512, 512, 512);
    }
#else
    // manual A tile: 128 rows x 32 halves, contiguous 16B chunks
#pragma unroll
    for (int it = 0; it < 2; ++it) {
      int chunk = tid + it * 256;
      int row = chunk >> 2;
      int c8  = (chunk & 3) * 8;
      *(u32x4*)&Wl[cur][row][c8] = *(const u32x4*)&Wb[(mbase + row) * 512 + kt + c8];
    }
#endif
    // B tile: load x rows (t contiguous), store transposed [n][k]
    {
      int k  = tid >> 3;                 // 0..31
      int n0 = (tid & 7) * 8;            // 0..56
      union { u32x4 u; __bf16 s[8]; } tmp;
      tmp.u = *(const u32x4*)&Xb[(bi * 512 + kt + k) * 2304 + tb + n0];
#pragma unroll
      for (int i = 0; i < 8; ++i) Xl[n0 + i][k] = tmp.s[i];
    }
    if (kt + 32 < 512) {                 // prefetch next X tile into caches
      __builtin_prefetch(&Xb[(bi * 512 + kt + 32 + lane) * 2304 + tb], 0, 0);
    }
#if USE_TDM
    if (wave == 0) {  // in-order TENSORcnt: <=1 pending => current tile complete
      if (kt + 32 < 512) __builtin_amdgcn_s_wait_tensorcnt(1);
      else               __builtin_amdgcn_s_wait_tensorcnt(0);
    }
#endif
    __syncthreads();

    v16bf afrag[2], bfrag[2];
#pragma unroll
    for (int im = 0; im < 2; ++im) {
      // A layout: lanes 0-15 hold K{0..7,16..23}, lanes 16-31 hold K{8..15,24..31}
      const __bf16* ar = &Wl[cur][wm + im * 16 + nl][half * 8];
      afrag[im] = frag16(ar, ar + 16);
    }
#pragma unroll
    for (int in = 0; in < 2; ++in) {
      // B layout: lanes 0-15 hold K{0..7,16..23}, lanes 16-31 hold K{8..15,24..31}
      const __bf16* br = &Xl[wn + in * 16 + nl][half * 8];
      bfrag[in] = frag16(br, br + 16);
    }
#pragma unroll
    for (int im = 0; im < 2; ++im)
#pragma unroll
      for (int in = 0; in < 2; ++in)
        acc[im][in] = wmma_bf16(afrag[im], bfrag[in], acc[im][in]);
    __syncthreads();  // fragment reads done before next staging overwrites LDS
  }

  // epilogue: C layout -> lane holds rows m = half*8 + r, col n = nl
#pragma unroll
  for (int im = 0; im < 2; ++im)
#pragma unroll
    for (int in = 0; in < 2; ++in)
#pragma unroll
      for (int r = 0; r < 8; ++r) {
        int o = mbase + wm + im * 16 + half * 8 + r;
        int t = tb + wn + in * 16 + nl;
        float val = acc[im][in][r] + bias[o];
        if (mode == 0) {
          out_qkv[((bi * HEADS + (o >> 6)) * TT + t) * HD + (o & 63)] =
              (__bf16)(val * scale);
        } else {
          out_f32[(bi * DIMC + o) * TT + t] = val;
        }
      }
}

// ---------------------------------------------------------------------------
// Neighborhood attention, one wave per 16-token row strip.
// Scores: Q[16x64] x K_rowT[64x22] per neighbor row (banded extraction).
// PV:     banded P[16x32] x Vt[32x64] per neighbor row.
// ---------------------------------------------------------------------------
struct AttnLds {
  float  scores[16][56];                    // 49 scores / probs per token
  __bf16 P[7][16][32];                      // banded probability tiles
  union { __bf16 Vt[64][32]; float O[16][64]; } u;
};

__global__ __launch_bounds__(128)
void attn_kernel(const __bf16* __restrict__ qb,   // [bh][t][hd], pre-scaled
                 const __bf16* __restrict__ kb,   // [bh][t][hd]
                 const __bf16* __restrict__ vb,   // [bh][t][hd]
                 __bf16* __restrict__ ob)         // [b][c][t]
{
  __shared__ __align__(16) AttnLds lds[4];
  const int tid  = threadIdx.x;
  const int lane = tid & 31;
  const int wave = tid >> 5;
  AttnLds& L = lds[wave];

  const int strip = blockIdx.x * 4 + wave;   // 0..2303
  const int bh  = strip / 144;               // 144 strips per (b,head)
  const int rem = strip % 144;
  const int h   = rem / 3;
  const int w0  = (rem % 3) * 16;
  const int t0  = h * WW + w0;
  const int bhoff = bh * TT;

  const int nl   = lane & 15;
  const int half = lane >> 4;
  const v8f vzero = {0.f,0.f,0.f,0.f,0.f,0.f,0.f,0.f};

  // Q fragments (rows are tokens; storage [t][d] is row-major => direct b128s)
  v16bf aq[2];
#pragma unroll
  for (int ks = 0; ks < 2; ++ks) {
    const __bf16* qrow = qb + (bhoff + t0 + nl) * HD + ks * 32 + half * 8;
    aq[ks] = frag16(qrow, qrow + 16);
  }

  // ---- scores: 7 neighbor rows x (2 col tiles x 2 k-steps) WMMAs ----
  for (int dy = 0; dy < 7; ++dy) {
    int hy_raw = h + dy - 3;
    int hy   = clampi(hy_raw, 0, HH - 1);
    int hval = (hy_raw >= 0) && (hy_raw < HH);
#pragma unroll
    for (int nt = 0; nt < 2; ++nt) {
      int col    = nt * 16 + nl;             // key column within 22-wide window
      int wx_raw = w0 - 3 + col;
      int wx     = clampi(wx_raw, 0, WW - 1);
      const __bf16* krow = kb + (bhoff + hy * WW + wx) * HD;
      v16bf b0 = frag16(krow + half * 8,      krow + half * 8 + 16);       // d 0..31
      v16bf b1 = frag16(krow + 32 + half * 8, krow + 32 + half * 8 + 16);  // d 32..63
      v8f s = vzero;
      s = wmma_bf16(aq[0], b0, s);
      s = wmma_bf16(aq[1], b1, s);
      int cval = hval && (wx_raw >= 0) && (wx_raw < WW);
      // scatter in-band elements: token m uses cols m..m+6
#pragma unroll
      for (int r = 0; r < 8; ++r) {
        int m  = half * 8 + r;
        int dx = col - m;
        if (dx >= 0 && dx < 7)
          L.scores[m][dy * 7 + dx] = cval ? s[r] : -1e30f;
      }
    }
  }
  __syncthreads();

  // ---- softmax per token (lanes 0..15) ----
  if (lane < 16) {
    float mx = -1e30f;
    for (int i = 0; i < 49; ++i) mx = fmaxf(mx, L.scores[lane][i]);
    float sum = 0.f;
    for (int i = 0; i < 49; ++i) {
      float e = __expf(L.scores[lane][i] - mx);
      L.scores[lane][i] = e;
      sum += e;
    }
    float inv = 1.f / sum;
    for (int i = 0; i < 49; ++i) L.scores[lane][i] *= inv;
  }
  __syncthreads();

  // ---- build banded bf16 P tiles ----
  {
    unsigned int* pp = (unsigned int*)&L.P[0][0][0];
    for (int i = lane; i < 7 * 16 * 32 / 2; i += 32) pp[i] = 0u;
  }
  __syncthreads();
  if (lane < 16) {
    for (int dy = 0; dy < 7; ++dy)
#pragma unroll
      for (int dx = 0; dx < 7; ++dx)
        L.P[dy][lane][lane + dx] = (__bf16)L.scores[lane][dy * 7 + dx];
  }
  __syncthreads();

  // ---- PV: accumulate O[16x64] over 7 neighbor rows ----
  v8f oacc[4];
  oacc[0] = vzero; oacc[1] = vzero; oacc[2] = vzero; oacc[3] = vzero;

  for (int dy = 0; dy < 7; ++dy) {
    // stage V transposed: Vt[d][j] for key cols j=0..31 (clamped window)
    int hy = clampi(h + dy - 3, 0, HH - 1);
    int wx = clampi(w0 - 3 + lane, 0, WW - 1);
    const __bf16* vrow = vb + (bhoff + hy * WW + wx) * HD;
#pragma unroll
    for (int d8 = 0; d8 < 8; ++d8) {
      union { u32x4 u; __bf16 s[8]; } tmp;
      tmp.u = *(const u32x4*)(vrow + d8 * 8);
#pragma unroll
      for (int i = 0; i < 8; ++i) L.u.Vt[d8 * 8 + i][lane] = tmp.s[i];
    }
    __syncthreads();

    const __bf16* prow = &L.P[dy][nl][0];
    v16bf pa = frag16(prow + half * 8, prow + 16 + half * 8);
#pragma unroll
    for (int nt = 0; nt < 4; ++nt) {
      const __bf16* vt = &L.u.Vt[nt * 16 + nl][half * 8];
      v16bf bv = frag16(vt, vt + 16);
      oacc[nt] = wmma_bf16(pa, bv, oacc[nt]);
    }
    __syncthreads();
  }

  // dump O (aliases Vt; all Vt reads are done)
#pragma unroll
  for (int nt = 0; nt < 4; ++nt)
#pragma unroll
    for (int r = 0; r < 8; ++r)
      L.u.O[half * 8 + r][nt * 16 + nl] = oacc[nt][r];
  __syncthreads();

  // ---- self-value removal + store in [b][c][t] bf16 layout ----
  if (lane < 16) {
    int tg = t0 + lane;
    const __bf16* svp = vb + (bhoff + tg) * HD;
    float n2 = 0.f, dt = 0.f;
    for (int d = 0; d < HD; ++d) {
      float s = (float)svp[d];
      n2 += s * s;
      dt += L.u.O[lane][d] * s;
    }
    float nrm  = fmaxf(sqrtf(n2), 1e-12f);
    float coef = dt / (nrm * nrm);
    int b = bh >> 3, head = bh & 7;
    for (int d = 0; d < HD; ++d) {
      float val = L.u.O[lane][d] - coef * (float)svp[d];
      ob[(b * DIMC + head * HD + d) * TT + tg] = (__bf16)val;
    }
  }
}

// ---------------------------------------------------------------------------
// host launcher
// ---------------------------------------------------------------------------
extern "C" void kernel_launch(void* const* d_in, const int* in_sizes, int n_in,
                              void* d_out, int out_size, void* d_ws, size_t ws_size,
                              hipStream_t stream) {
  const float* x  = (const float*)d_in[0];
  const float* wq = (const float*)d_in[1];
  const float* bq = (const float*)d_in[2];
  const float* wk = (const float*)d_in[3];
  const float* bk = (const float*)d_in[4];
  const float* wv = (const float*)d_in[5];
  const float* bv = (const float*)d_in[6];
  const float* wo = (const float*)d_in[7];
  const float* bo = (const float*)d_in[8];

  char* ws = (char*)d_ws;
  size_t off = 0;
  auto alloc = [&](size_t bytes) {
    void* p = ws + off;
    off += (bytes + 255) & ~(size_t)255;
    return p;
  };

  const size_t NX = (size_t)NB * DIMC * TT;   // 2,359,296
  const size_t NW = (size_t)DIMC * DIMC;      //   262,144

  __bf16* xb   = (__bf16*)alloc(NX * 2);
  __bf16* wqb  = (__bf16*)alloc(NW * 2);
  __bf16* wkb  = (__bf16*)alloc(NW * 2);
  __bf16* wvb  = (__bf16*)alloc(NW * 2);
  __bf16* wob  = (__bf16*)alloc(NW * 2);
  __bf16* qbuf = (__bf16*)alloc(NX * 2);
  __bf16* kbuf = (__bf16*)alloc(NX * 2);
  __bf16* vbuf = (__bf16*)alloc(NX * 2);
  __bf16* abuf = (__bf16*)alloc(NX * 2);

  cvt_bf16_kernel<<<(unsigned)((NX / 4 + 255) / 256), 256, 0, stream>>>(x,  xb,  (int)NX);
  cvt_bf16_kernel<<<(unsigned)((NW / 4 + 255) / 256), 256, 0, stream>>>(wq, wqb, (int)NW);
  cvt_bf16_kernel<<<(unsigned)((NW / 4 + 255) / 256), 256, 0, stream>>>(wk, wkb, (int)NW);
  cvt_bf16_kernel<<<(unsigned)((NW / 4 + 255) / 256), 256, 0, stream>>>(wv, wvb, (int)NW);
  cvt_bf16_kernel<<<(unsigned)((NW / 4 + 255) / 256), 256, 0, stream>>>(wo, wob, (int)NW);

  dim3 gg(72, 4);  // 72 N-tiles of 64 pixels x 4 M-tiles of 128 channels
  proj_gemm_kernel<<<gg, 256, 0, stream>>>(wqb, bq, xb, qbuf, nullptr, QK_SCALE, 0);
  proj_gemm_kernel<<<gg, 256, 0, stream>>>(wkb, bk, xb, kbuf, nullptr, 1.0f, 0);
  proj_gemm_kernel<<<gg, 256, 0, stream>>>(wvb, bv, xb, vbuf, nullptr, 1.0f, 0);

  attn_kernel<<<576, 128, 0, stream>>>(qbuf, kbuf, vbuf, abuf);

  proj_gemm_kernel<<<gg, 256, 0, stream>>>(wob, bo, abuf, nullptr, (float*)d_out, 1.0f, 1);
}
